// GPT3ParallelSelfAttention_58385785422450
// MI455X (gfx1250) — matrix operand order repaired and checked by reading, not compiled, not run on hardware
//
#include <hip/hip_runtime.h>

// ---------------------------------------------------------------------------
// GPT-3 self-attention for MI455X (gfx1250, wave32, WMMA bf16 -> f32 accum)
//   B=1, S=4096, H=1024, N=16 heads, head dim D=64
// Pipeline (all on `stream`):
//   1) convert X, W_qkv, W_dense  f32 -> bf16
//   2) QKV GEMM (+bias) with async-to-LDS double buffering, scatter into
//      Q[n][s][d], K[n][s][d], Vt[n][d][s] bf16
//   3) flash-attention per (head, 16-query tile): online softmax, causal mask
//      computed analytically, WMMA for QK^T and P*V, prefetch next K block
//   4) dense GEMM (+bias, async-to-LDS double buffered) -> f32 output
// ---------------------------------------------------------------------------

typedef __attribute__((ext_vector_type(16))) __bf16 v16bf;
typedef __attribute__((ext_vector_type(8)))  float  v8f;

constexpr int S  = 4096;
constexpr int H  = 1024;
constexpr int NH = 16;
constexpr int D  = 64;
constexpr int O3 = 3 * H;   // 3072

constexpr int BM = 128;     // block M tile (8 waves x 16 rows)
constexpr int BN = 64;      // block N tile
constexpr int BK = 32;      // K step (one WMMA K)

__device__ __forceinline__ unsigned short f32_to_bf16_bits(float f) {
  unsigned int u = __float_as_uint(f);
  unsigned int rnd = 0x7FFFu + ((u >> 16) & 1u);   // round-to-nearest-even
  return (unsigned short)((u + rnd) >> 16);
}

union FragU {
  v16bf v;
  unsigned short s[16];
  uint4 q[2];
};

// load 8 bf16 from p0 into elements 0..7 and 8 bf16 from p1 into 8..15
__device__ __forceinline__ v16bf load2x8(const unsigned short* p0,
                                         const unsigned short* p1) {
  FragU u;
  u.q[0] = *reinterpret_cast<const uint4*>(p0);
  u.q[1] = *reinterpret_cast<const uint4*>(p1);
  return u.v;
}
__device__ __forceinline__ v16bf load16(const unsigned short* p) {
  return load2x8(p, p + 8);
}

__device__ __forceinline__ v8f wmma_bf16(v16bf a, v16bf b, v8f c) {
  return __builtin_amdgcn_wmma_f32_16x16x32_bf16(
      /*neg_a=*/false, a, /*neg_b=*/false, b,
      /*c_mod=*/(short)0, c, /*reuse_a=*/false, /*reuse_b=*/false);
}

// ---- CDNA5 async copy global -> LDS (ASYNCcnt), via inline asm -------------
// LDS address operand: low 32 bits of the generic pointer == LDS byte offset.
__device__ __forceinline__ void async_b128(const unsigned short* g, void* l) {
  unsigned laddr = (unsigned)(uintptr_t)l;
  unsigned long long gaddr = (unsigned long long)(uintptr_t)g;
  asm volatile("global_load_async_to_lds_b128 %0, %1, off"
               :: "v"(laddr), "v"(gaddr) : "memory");
}
__device__ __forceinline__ void wait_async0() {
  asm volatile("s_wait_asynccnt 0x0" ::: "memory");
}

// ---------------------------------------------------------------------------
// f32 -> bf16 conversion
// ---------------------------------------------------------------------------
__global__ void cvt_bf16(const float* __restrict__ src,
                         unsigned short* __restrict__ dst, int n) {
  int i = blockIdx.x * blockDim.x + threadIdx.x;
  if (i < n) dst[i] = f32_to_bf16_bits(src[i]);
}

// ---------------------------------------------------------------------------
// QKV GEMM: C[S x 3H] = X[S x H] * W^T (+bias), scatter to Q / K / Vt (bf16)
// Block = 8 waves, tile BM=128 x BN=64, async-to-LDS double buffered over BK=32.
// ---------------------------------------------------------------------------
__global__ __launch_bounds__(256) void qkv_gemm(
    const unsigned short* __restrict__ X,
    const unsigned short* __restrict__ W,
    const float* __restrict__ bias,
    unsigned short* __restrict__ Qb,
    unsigned short* __restrict__ Kb,
    unsigned short* __restrict__ Vtb) {
  __shared__ unsigned short At[2][BM * BK];   // 2 x 8 KB
  __shared__ unsigned short Bt[2][BN * BK];   // 2 x 4 KB

  const int tid  = threadIdx.x;
  const int lane = tid & 31;
  const int wv   = tid >> 5;            // wave 0..7 -> m subtile
  const int col  = lane & 15;
  const int hi   = lane >> 4;

  const int tiles_n = O3 / BN;          // 48
  const int mblk = (blockIdx.x / tiles_n) * BM;
  const int nblk = (blockIdx.x % tiles_n) * BN;

  auto stage = [&](int buf, int k0) {
    // A tile: 128 rows x 32 k -> 512 x 16B chunks, 2 per thread
#pragma unroll
    for (int i = 0; i < 2; ++i) {
      const int c = tid + 256 * i;
      const int r = c >> 2, q = c & 3;
      async_b128(X + (mblk + r) * H + k0 + q * 8, &At[buf][c * 8]);
    }
    // B tile: 64 rows x 32 k -> 256 x 16B chunks, 1 per thread
    {
      const int r = tid >> 2, q = tid & 3;
      async_b128(W + (nblk + r) * H + k0 + q * 8, &Bt[buf][tid * 8]);
    }
  };

  v8f acc[4] = {};
  stage(0, 0);
  wait_async0();
  __syncthreads();

  const int KSTEPS = H / BK;            // 32
  for (int kb = 0; kb < KSTEPS; ++kb) {
    const int buf = kb & 1;
    if (kb + 1 < KSTEPS) stage(buf ^ 1, (kb + 1) * BK);

    const unsigned short* arow = &At[buf][(wv * 16 + col) * BK];
    const v16bf a = load16(arow + 16 * hi);
#pragma unroll
    for (int t = 0; t < 4; ++t) {
      const unsigned short* brow = &Bt[buf][(t * 16 + col) * BK + 16 * hi];
      acc[t] = wmma_bf16(a, load16(brow), acc[t]);
    }
    wait_async0();                      // next-stage async loads complete
    __syncthreads();
  }

  const int mbase = mblk + wv * 16;
#pragma unroll
  for (int t = 0; t < 4; ++t) {
    const int o  = nblk + t * 16 + col;
    const float bv = bias[o];
#pragma unroll
    for (int g = 0; g < 8; ++g) {
      const int srow = mbase + g + 8 * hi;     // C: row = g + 8*hi
      const unsigned short bits = f32_to_bf16_bits(acc[t][g] + bv);
      if (o < H) {                              // Q -> [n][s][d]
        const int n = o >> 6, d = o & 63;
        Qb[(n * S + srow) * D + d] = bits;
      } else if (o < 2 * H) {                   // K -> [n][s][d]
        const int oo = o - H, n = oo >> 6, d = oo & 63;
        Kb[(n * S + srow) * D + d] = bits;
      } else {                                  // V -> transposed [n][d][s]
        const int oo = o - 2 * H, n = oo >> 6, d = oo & 63;
        Vtb[(n * D + d) * S + srow] = bits;
      }
    }
  }
}

// ---------------------------------------------------------------------------
// Flash attention, one wave per (head, 16-query tile).
// Scores computed transposed (A = K rows, B = Q^T) so the S^T C-fragment
// layout coincides with the P A-fragment layout for P*V (no lane transpose).
// ---------------------------------------------------------------------------
__global__ __launch_bounds__(128) void attn(
    const unsigned short* __restrict__ Qb,
    const unsigned short* __restrict__ Kb,
    const unsigned short* __restrict__ Vtb,
    unsigned short* __restrict__ ctx) {
  const int wid  = blockIdx.x * (blockDim.x >> 5) + (threadIdx.x >> 5);
  const int lane = threadIdx.x & 31;
  const int col  = lane & 15;
  const int hi   = lane >> 4;

  const int qtiles = S / 16;                   // 256
  const int head  = wid / qtiles;
  const int qbase = (wid % qtiles) * 16;

  const unsigned short* Qn  = Qb  + head * S * D;
  const unsigned short* Kn  = Kb  + head * S * D;
  const unsigned short* Vtn = Vtb + head * D * S;

  // Q^T B-fragments for both 32-wide d halves (lane = query column)
  const unsigned short* qrow = Qn + (qbase + col) * D;
  const v16bf bq0 = load2x8(qrow + 0  + 8 * hi, qrow + 0  + 8 * hi + 16);
  const v16bf bq1 = load2x8(qrow + 32 + 8 * hi, qrow + 32 + 8 * hi + 16);

  v8f o0 = {}, o1 = {}, o2 = {}, o3 = {};
  float m = -1e30f, l = 0.0f;

  const int nkb = (qbase + 16 + 31) >> 5;      // 32-key blocks covering causal span
  for (int kb = 0; kb < nkb; ++kb) {
    const int k0 = kb * 32;

    if (kb + 1 < nkb) {                        // prefetch next K block rows
      __builtin_prefetch(Kn + (k0 + 32 + col) * D, 0, 0);
      __builtin_prefetch(Kn + (k0 + 48 + col) * D, 0, 0);
    }

    // S^T tiles: c0 = keys [k0,k0+16), c1 = keys [k0+16,k0+32); K-dim = d (2 halves)
    v8f c0 = {}, c1 = {};
    {
      const unsigned short* kr0 = Kn + (k0 + col) * D;
      c0 = wmma_bf16(load2x8(kr0 + 8 * hi,      kr0 + 8 * hi + 16),      bq0, c0);
      c0 = wmma_bf16(load2x8(kr0 + 32 + 8 * hi, kr0 + 32 + 8 * hi + 16), bq1, c0);
      const unsigned short* kr1 = Kn + (k0 + 16 + col) * D;
      c1 = wmma_bf16(load2x8(kr1 + 8 * hi,      kr1 + 8 * hi + 16),      bq0, c1);
      c1 = wmma_bf16(load2x8(kr1 + 32 + 8 * hi, kr1 + 32 + 8 * hi + 16), bq1, c1);
    }

    // causal mask + 1/sqrt(64) scale; reference: masked = exactly -10000
    const int q = qbase + col;
    float bm = -1e30f;
#pragma unroll
    for (int g = 0; g < 8; ++g) {
      const int key0 = k0 + g + 8 * hi;        // S^T: row (key) = g + 8*hi
      const float s0 = (key0      <= q) ? c0[g] * 0.125f : -10000.0f;
      const float s1 = (key0 + 16 <= q) ? c1[g] * 0.125f : -10000.0f;
      c0[g] = s0; c1[g] = s1;
      bm = fmaxf(bm, fmaxf(s0, s1));
    }
    bm = fmaxf(bm, __shfl_xor(bm, 16, 32));    // combine the two key halves
    const float mnew  = fmaxf(m, bm);
    const float alpha = __expf(m - mnew);

    // exp + pack P directly into A-fragment layout (no cross-lane movement)
    FragU pa;
    float psum = 0.0f;
#pragma unroll
    for (int g = 0; g < 8; ++g) {
      const float p0 = __expf(c0[g] - mnew);
      const float p1 = __expf(c1[g] - mnew);
      psum += p0 + p1;
      pa.s[g]     = f32_to_bf16_bits(p0);      // e<8  : keys k0 + e + 8*hi
      pa.s[g + 8] = f32_to_bf16_bits(p1);      // e>=8 : keys k0+16 + (e-8)+8*hi
    }
    psum += __shfl_xor(psum, 16, 32);
    l = l * alpha + psum;
    m = mnew;

    // rescale O rows: O row = g + 8*hi -> broadcast alpha from lane (8*hi + g)
#pragma unroll
    for (int g = 0; g < 8; ++g) {
      const float ag = __shfl(alpha, hi * 8 + g, 32);
      o0[g] *= ag; o1[g] *= ag; o2[g] *= ag; o3[g] *= ag;
    }

    // O += P * V ; B = V[32 keys x 16 d] from Vt (contiguous in s)
    o0 = wmma_bf16(pa.v, load2x8(Vtn + (0  + col) * S + k0 + 8 * hi, Vtn + (0  + col) * S + k0 + 8 * hi + 16), o0);
    o1 = wmma_bf16(pa.v, load2x8(Vtn + (16 + col) * S + k0 + 8 * hi, Vtn + (16 + col) * S + k0 + 8 * hi + 16), o1);
    o2 = wmma_bf16(pa.v, load2x8(Vtn + (32 + col) * S + k0 + 8 * hi, Vtn + (32 + col) * S + k0 + 8 * hi + 16), o2);
    o3 = wmma_bf16(pa.v, load2x8(Vtn + (48 + col) * S + k0 + 8 * hi, Vtn + (48 + col) * S + k0 + 8 * hi + 16), o3);
  }

  // normalize and store ctx[s][head*64 + d] as bf16
  const float linv = 1.0f / l;
#pragma unroll
  for (int g = 0; g < 8; ++g) {
    const float ng = __shfl(linv, hi * 8 + g, 32);
    const int srow = qbase + g + 8 * hi;
    unsigned short* crow = ctx + srow * H + head * D;
    crow[0  + col] = f32_to_bf16_bits(o0[g] * ng);
    crow[16 + col] = f32_to_bf16_bits(o1[g] * ng);
    crow[32 + col] = f32_to_bf16_bits(o2[g] * ng);
    crow[48 + col] = f32_to_bf16_bits(o3[g] * ng);
  }
}

// ---------------------------------------------------------------------------
// Dense projection: out[S x H] = ctx[S x H] * Wd^T + b  (f32 output)
// Same async-to-LDS double-buffered structure as qkv_gemm.
// ---------------------------------------------------------------------------
__global__ __launch_bounds__(256) void dense_gemm(
    const unsigned short* __restrict__ ctx,
    const unsigned short* __restrict__ Wd,
    const float* __restrict__ bias,
    float* __restrict__ out) {
  __shared__ unsigned short At[2][BM * BK];
  __shared__ unsigned short Bt[2][BN * BK];

  const int tid  = threadIdx.x;
  const int lane = tid & 31;
  const int wv   = tid >> 5;
  const int col  = lane & 15;
  const int hi   = lane >> 4;

  const int tiles_n = H / BN;           // 16
  const int mblk = (blockIdx.x / tiles_n) * BM;
  const int nblk = (blockIdx.x % tiles_n) * BN;

  auto stage = [&](int buf, int k0) {
#pragma unroll
    for (int i = 0; i < 2; ++i) {
      const int c = tid + 256 * i;
      const int r = c >> 2, q = c & 3;
      async_b128(ctx + (mblk + r) * H + k0 + q * 8, &At[buf][c * 8]);
    }
    {
      const int r = tid >> 2, q = tid & 3;
      async_b128(Wd + (nblk + r) * H + k0 + q * 8, &Bt[buf][tid * 8]);
    }
  };

  v8f acc[4] = {};
  stage(0, 0);
  wait_async0();
  __syncthreads();

  const int KSTEPS = H / BK;
  for (int kb = 0; kb < KSTEPS; ++kb) {
    const int buf = kb & 1;
    if (kb + 1 < KSTEPS) stage(buf ^ 1, (kb + 1) * BK);

    const unsigned short* arow = &At[buf][(wv * 16 + col) * BK];
    const v16bf a = load16(arow + 16 * hi);
#pragma unroll
    for (int t = 0; t < 4; ++t) {
      const unsigned short* brow = &Bt[buf][(t * 16 + col) * BK + 16 * hi];
      acc[t] = wmma_bf16(a, load16(brow), acc[t]);
    }
    wait_async0();
    __syncthreads();
  }

  const int mbase = mblk + wv * 16;
#pragma unroll
  for (int t = 0; t < 4; ++t) {
    const int o  = nblk + t * 16 + col;
    const float bv = bias[o];
#pragma unroll
    for (int g = 0; g < 8; ++g) {
      const int srow = mbase + g + 8 * hi;
      out[srow * H + o] = acc[t][g] + bv;
    }
  }
}

// ---------------------------------------------------------------------------
extern "C" void kernel_launch(void* const* d_in, const int* in_sizes, int n_in,
                              void* d_out, int out_size, void* d_ws, size_t ws_size,
                              hipStream_t stream) {
  const float* hs      = (const float*)d_in[0];   // [1,4096,1024]
  // d_in[1] = ltor_mask: ignored (causal mask computed analytically)
  const float* w_qkv   = (const float*)d_in[2];   // [3072,1024]
  const float* b_qkv   = (const float*)d_in[3];   // [3072]
  const float* w_dense = (const float*)d_in[4];   // [1024,1024]
  const float* b_dense = (const float*)d_in[5];   // [1024]
  float* out = (float*)d_out;

  char* ws = (char*)d_ws;
  unsigned short* Xbf  = (unsigned short*)ws;  ws += (size_t)S * H  * 2;   // 8 MB
  unsigned short* Wqb  = (unsigned short*)ws;  ws += (size_t)O3 * H * 2;   // 6 MB
  unsigned short* Wdb  = (unsigned short*)ws;  ws += (size_t)H * H  * 2;   // 2 MB
  unsigned short* Qb   = (unsigned short*)ws;  ws += (size_t)S * H  * 2;   // 8 MB
  unsigned short* Kb   = (unsigned short*)ws;  ws += (size_t)S * H  * 2;   // 8 MB
  unsigned short* Vtb  = (unsigned short*)ws;  ws += (size_t)S * H  * 2;   // 8 MB
  unsigned short* ctxb = (unsigned short*)ws;  ws += (size_t)S * H  * 2;   // 8 MB

  // 1) f32 -> bf16 conversions
  {
    int n = S * H;
    cvt_bf16<<<(n + 255) / 256, 256, 0, stream>>>(hs, Xbf, n);
    n = O3 * H;
    cvt_bf16<<<(n + 255) / 256, 256, 0, stream>>>(w_qkv, Wqb, n);
    n = H * H;
    cvt_bf16<<<(n + 255) / 256, 256, 0, stream>>>(w_dense, Wdb, n);
  }

  // 2) QKV GEMM: (4096/128)*(3072/64) = 1536 blocks of 8 waves
  qkv_gemm<<<(S / BM) * (O3 / BN), 256, 0, stream>>>(Xbf, Wqb, b_qkv,
                                                     Qb, Kb, Vtb);

  // 3) attention: 16 heads * 256 q-tiles = 4096 waves, 4 waves/block
  attn<<<NH * (S / 16) / 4, 128, 0, stream>>>(Qb, Kb, Vtb, ctxb);

  // 4) dense projection: (4096/128)*(1024/64) = 512 blocks of 8 waves
  dense_gemm<<<(S / BM) * (H / BN), 256, 0, stream>>>(ctxb, Wdb, b_dense, out);
}
